// ST_LSTM_21534966022555
// MI455X (gfx1250) — matrix operand order repaired and checked by reading, not compiled, not run on hardware
//
#include <hip/hip_runtime.h>
#include <hip/hip_bf16.h>
#include <cstddef>

// Problem constants (from reference)
#define Bb     64
#define T1v    49
#define NBONE  24
#define HID    512
#define DD     3
#define SSv    25
#define NGATE  (5*HID)     // 2560
#define KT0    1056        // 32 (padded x) + 512 + 512
#define KT1    1536        // 512 + 512 + 512
#define PREC_BASE ((size_t)Bb*SSv*NBONE*DD)   // 115200 floats: start of pre_c in d_out

typedef __attribute__((ext_vector_type(16))) __bf16 v16bf;
typedef __attribute__((ext_vector_type(8)))  __bf16 v8bf;
typedef __attribute__((ext_vector_type(8)))  float  v8f;

__device__ __forceinline__ float sigf(float x) { return 1.0f / (1.0f + __expf(-x)); }

// ---------------------------------------------------------------------------
// One-time: fp32 weights -> transposed bf16  WT[bone][n=2560][k=KT]
// ---------------------------------------------------------------------------
__global__ void wconv(const float* __restrict__ U, const float* __restrict__ Wt,
                      const float* __restrict__ Ws, __bf16* __restrict__ out,
                      int uRows, int xK, int KT) {
  size_t tid = (size_t)blockIdx.x * blockDim.x + threadIdx.x;
  size_t total = (size_t)NBONE * NGATE * KT;
  if (tid >= total) return;
  int k = (int)(tid % KT);
  size_t t = tid / KT;
  int n = (int)(t % NGATE);
  int bone = (int)(t / NGATE);
  float v;
  if (k < xK)            v = (k < uRows) ? U[((size_t)bone * uRows + k) * NGATE + n] : 0.0f;
  else if (k < xK + HID) v = Wt[((size_t)bone * HID + (k - xK)) * NGATE + n];
  else                   v = Ws[((size_t)bone * HID + (k - xK - HID)) * NGATE + n];
  out[tid] = (__bf16)v;
}

// ---------------------------------------------------------------------------
// One-time: initial grid states (parity-0 buffers), cols 1..NB
// ---------------------------------------------------------------------------
__global__ void init_states(const float* __restrict__ hs, const float* __restrict__ cs,
                            const float* __restrict__ gt,
                            __bf16* __restrict__ h0g, float* __restrict__ c0g,
                            __bf16* __restrict__ h1g, float* __restrict__ c1g) {
  size_t tid = (size_t)blockIdx.x * blockDim.x + threadIdx.x;
  if (tid >= (size_t)Bb * NBONE * HID) return;
  int h = (int)(tid % HID);
  size_t t = tid / HID;
  int bone = (int)(t % NBONE);
  int b = (int)(t / NBONE);
  float sh = 0.0f, sc = 0.0f;
  for (int tt = 0; tt < T1v; ++tt) {
    size_t src = (((size_t)b * T1v + tt) * NBONE + bone) * HID + h;
    sh += hs[src];
    sc += cs[src];
  }
  float hmean = sh * (1.0f / T1v);
  float cmean = sc * (1.0f / T1v);
  float h1v = (gt[((size_t)b * NBONE + bone) * HID + h] + sh) * (1.0f / (T1v + 1));
  size_t dst = ((size_t)(bone + 1) * Bb + b) * HID + h;
  h0g[dst] = (__bf16)hmean;
  c0g[dst] = cmean;
  c1g[dst] = cmean;
  h1g[dst] = (__bf16)h1v;
}

// One-time: frame-0 input p[:,0] -> padded bf16 x buffer (parity 0)
__global__ void init_x(const float* __restrict__ p, __bf16* __restrict__ x0) {
  int tid = blockIdx.x * blockDim.x + threadIdx.x;
  if (tid >= NBONE * Bb * DD) return;
  int d = tid % DD;
  int t = tid / DD;
  int b = t % Bb;
  int bone = t / Bb;
  x0[((size_t)bone * Bb + b) * 32 + d] =
      (__bf16)p[(((size_t)b * T1v + 0) * NBONE + bone) * DD + d];
}

// ---------------------------------------------------------------------------
// GEMM segment: STEPS k-chunks of 32 from one uniform A source.
// A layout used here (16-bit A 16x32, contiguous k, not the ISA table order): lane l, row M=l&15:
//   16 contiguous K at 16*(l>>4)  (two 16B loads)
// B layout (32x16): lane l, col N=l&15, 16 contiguous K at 16*(l>>4)
//   (weights pre-transposed so a W column is contiguous).
// ---------------------------------------------------------------------------
template <int STEPS>
__device__ __forceinline__ void gemm_seg(
    const __bf16* __restrict__ arow,   // A row base (+half*16 applied)
    const __bf16* __restrict__ wbase,  // gate-0 weight base (+half*16 applied)
    size_t gateStride,                 // HID*KT
    v8f& c0, v8f& c1, v8f& c2, v8f& c3, v8f& c4) {
#pragma unroll 2
  for (int kc = 0; kc < STEPS; ++kc) {
    const __bf16* ap = arow + kc * 32;
    v8bf alo = *(const v8bf*)(ap);
    v8bf ahi = *(const v8bf*)(ap + 8);
    v16bf a;
#pragma unroll
    for (int i = 0; i < 8; ++i) { a[i] = alo[i]; a[i + 8] = ahi[i]; }

    const __bf16* wp = wbase + kc * 32;
    v16bf b0 = *(const v16bf*)(wp);
    v16bf b1 = *(const v16bf*)(wp + gateStride);
    v16bf b2 = *(const v16bf*)(wp + 2 * gateStride);
    v16bf b3 = *(const v16bf*)(wp + 3 * gateStride);
    v16bf b4 = *(const v16bf*)(wp + 4 * gateStride);

    c0 = __builtin_amdgcn_wmma_f32_16x16x32_bf16(false, a, false, b0, (short)0, c0, false, false);
    c1 = __builtin_amdgcn_wmma_f32_16x16x32_bf16(false, a, false, b1, (short)0, c1, false, false);
    c2 = __builtin_amdgcn_wmma_f32_16x16x32_bf16(false, a, false, b2, (short)0, c2, false, false);
    c3 = __builtin_amdgcn_wmma_f32_16x16x32_bf16(false, a, false, b3, (short)0, c3, false, false);
    c4 = __builtin_amdgcn_wmma_f32_16x16x32_bf16(false, a, false, b4, (short)0, c4, false, false);
  }
}

// ---------------------------------------------------------------------------
// One ST-LSTM cell:  Z[64,2560] = [X | Ht | Hs] @ WT^T + b  -> gates -> h,c
// XSTEPS = 1 (layer 0, x padded to 32) or 16 (layer 1, x = 512 hidden).
// 128 waves total: 32 blocks x 4 waves; waves of a block share the M-tile
// (A rows duplicated in cache) and span 4 consecutive N-tiles (unique weights).
// ---------------------------------------------------------------------------
template <int XSTEPS>
__global__ __launch_bounds__(128) void st_cell(
    const __bf16* __restrict__ WT,   // [2560][KT] bf16 (this bone)
    const float*  __restrict__ bias, // [2560] fp32 (this bone)
    const __bf16* __restrict__ X,    // [64][xStride] bf16
    const __bf16* __restrict__ Ht,   // [64][512] bf16 (temporal state, prev frame)
    const __bf16* __restrict__ Hs,   // [64][512] bf16 (spatial state, prev bone)
    const float*  __restrict__ Ct,   // [64][512] f32
    const float*  __restrict__ Cs,   // [64][512] f32
    __bf16* __restrict__ HgOut,      // next-frame grid slot (bf16)
    __bf16* __restrict__ HcOut,      // next-bone column slot (bf16)
    float*  __restrict__ CgOut,      // next-frame grid slot (f32)
    float*  __restrict__ CcOut,      // next-bone column slot (f32)
    float*  __restrict__ preC,       // optional d_out pre_c slot (layer 1)
    int preCStride) {
  constexpr int KT = XSTEPS * 32 + 2 * HID;
  constexpr int xStride = (XSTEPS == 1) ? 32 : HID;
  constexpr size_t GS = (size_t)HID * KT;      // gate stride in WT

  const int lane  = threadIdx.x & 31;
  const int wv    = threadIdx.x >> 5;          // wave in block: 0..3
  const int mtile = blockIdx.x & 3;            // shared by the block's 4 waves
  const int ntile = ((blockIdx.x >> 2) << 2) | wv;   // 0..31
  const int half  = lane >> 4;
  const int l15   = lane & 15;
  const int arow  = mtile * 16 + l15;          // batch row for A
  const int ncol  = ntile * 16 + l15;          // hidden column (within a gate)

  v8f acc0 = {}, acc1 = {}, acc2 = {}, acc3 = {}, acc4 = {};

  const __bf16* wcol  = WT + (size_t)ncol * KT + half * 16;  // gate-0 column
  const __bf16* xrow  = X  + (size_t)arow * xStride + half * 16;
  const __bf16* htrow = Ht + ((size_t)arow << 9) + half * 16;
  const __bf16* hsrow = Hs + ((size_t)arow << 9) + half * 16;

  gemm_seg<XSTEPS>(xrow,  wcol,                    GS, acc0, acc1, acc2, acc3, acc4);
  gemm_seg<16>(    htrow, wcol + XSTEPS * 32,      GS, acc0, acc1, acc2, acc3, acc4);
  gemm_seg<16>(    hsrow, wcol + XSTEPS * 32 + HID, GS, acc0, acc1, acc2, acc3, acc4);

  // Gate order from reference split: [i, f_s, f_t, o, g]
  const float bi  = bias[0 * HID + ncol];
  const float bfs = bias[1 * HID + ncol];
  const float bft = bias[2 * HID + ncol];
  const float bo  = bias[3 * HID + ncol];
  const float bg  = bias[4 * HID + ncol];

#pragma unroll
  for (int r = 0; r < 8; ++r) {
    const int m = mtile * 16 + half * 8 + r;   // C/D layout: lanes>=16 hold M=r+8
    const size_t idx = ((size_t)m << 9) + ncol;
    const float ct = Ct[idx];
    const float cs = Cs[idx];
    const float zi  = acc0[r] + bi;
    const float zfs = acc1[r] + bfs;
    const float zft = acc2[r] + bft;
    const float zo  = acc3[r] + bo;
    const float zg  = acc4[r] + bg;
    const float cn = sigf(zi) * tanhf(zg) + sigf(zft) * ct + sigf(zfs) * cs;
    const float hh = sigf(zo) * tanhf(cn);
    CgOut[idx] = cn;
    CcOut[idx] = cn;
    const __bf16 hb = (__bf16)hh;
    HgOut[idx] = hb;
    HcOut[idx] = hb;
    if (preC) preC[(size_t)m * preCStride + ncol] = cn;
  }
}

// ---------------------------------------------------------------------------
// Output projection: pre[b,f,bone,:] = h1[bone+1] @ W_out + b_out + x[bone23]
// Also writes the next frame's bf16 x input (parity-flipped buffer).
// ---------------------------------------------------------------------------
__global__ void proj(const __bf16* __restrict__ h1col, const float* __restrict__ Wout,
                     const float* __restrict__ bout, const __bf16* __restrict__ xcur,
                     __bf16* __restrict__ xnext, float* __restrict__ out, int f) {
  int tid = blockIdx.x * blockDim.x + threadIdx.x;
  if (tid >= Bb * NBONE * DD) return;
  int d = tid % DD;
  int t = tid / DD;
  int bone = t % NBONE;
  int b = t / NBONE;
  const __bf16* hp = h1col + ((size_t)(bone + 1) * Bb + b) * HID;
  float acc = bout[d];
  for (int k = 0; k < HID; ++k) acc += (float)hp[k] * Wout[k * DD + d];
  acc += (float)xcur[((size_t)(NBONE - 1) * Bb + b) * 32 + d];  // input of LAST bone
  out[(((size_t)b * SSv + f) * NBONE + bone) * DD + d] = acc;
  xnext[((size_t)bone * Bb + b) * 32 + d] = (__bf16)acc;        // feeds next frame
}

// ---------------------------------------------------------------------------
extern "C" void kernel_launch(void* const* d_in, const int* in_sizes, int n_in,
                              void* d_out, int out_size, void* d_ws, size_t ws_size,
                              hipStream_t stream) {
  (void)in_sizes; (void)n_in; (void)out_size; (void)ws_size;
  const float* hidden = (const float*)d_in[0];
  const float* cells  = (const float*)d_in[1];
  const float* gts    = (const float*)d_in[2];
  // d_in[3] = global_s_state: unused by the reference computation
  const float* p      = (const float*)d_in[4];
  const float* U0  = (const float*)d_in[5];
  const float* Wt0 = (const float*)d_in[6];
  const float* Ws0 = (const float*)d_in[7];
  const float* b0  = (const float*)d_in[8];
  const float* U1  = (const float*)d_in[9];
  const float* Wt1 = (const float*)d_in[10];
  const float* Ws1 = (const float*)d_in[11];
  const float* b1  = (const float*)d_in[12];
  const float* Wout = (const float*)d_in[13];
  const float* bout = (const float*)d_in[14];
  float* out = (float*)d_out;

  // Workspace carve-up (~318 MB bf16 weights + ~30 MB state)
  size_t off = 0;
  auto take = [&](size_t bytes) -> char* {
    char* ptr = (char*)d_ws + off;
    off += (bytes + 255) & ~(size_t)255;
    return ptr;
  };
  __bf16* WT0 = (__bf16*)take((size_t)NBONE * NGATE * KT0 * sizeof(__bf16));
  __bf16* WT1 = (__bf16*)take((size_t)NBONE * NGATE * KT1 * sizeof(__bf16));
  const size_t slab = (size_t)(NBONE + 1) * Bb * HID;   // one [NB+1][B][H] plane
  size_t stateStart = off;
  __bf16* h0g = (__bf16*)take(2 * slab * sizeof(__bf16));   // double-buffered grids
  __bf16* h1g = (__bf16*)take(2 * slab * sizeof(__bf16));
  float*  c0g = (float*) take(2 * slab * sizeof(float));
  float*  c1g = (float*) take(2 * slab * sizeof(float));
  __bf16* h0c = (__bf16*)take(slab * sizeof(__bf16));       // per-frame bone columns
  __bf16* h1c = (__bf16*)take(slab * sizeof(__bf16));
  float*  c0c = (float*) take(slab * sizeof(float));
  float*  c1c = (float*) take(slab * sizeof(float));
  __bf16* xall = (__bf16*)take((size_t)2 * NBONE * Bb * 32 * sizeof(__bf16));

  // One-time weight convert+transpose to bf16
  {
    size_t tot0 = (size_t)NBONE * NGATE * KT0;
    wconv<<<(unsigned)((tot0 + 255) / 256), 256, 0, stream>>>(U0, Wt0, Ws0, WT0, DD, 32, KT0);
    size_t tot1 = (size_t)NBONE * NGATE * KT1;
    wconv<<<(unsigned)((tot1 + 255) / 256), 256, 0, stream>>>(U1, Wt1, Ws1, WT1, HID, HID, KT1);
  }
  // Zero all state (grid col 0 / x pads stay zero forever)
  hipMemsetAsync((char*)d_ws + stateStart, 0, off - stateStart, stream);
  init_states<<<(Bb * NBONE * HID + 255) / 256, 256, 0, stream>>>(hidden, cells, gts,
                                                                  h0g, c0g, h1g, c1g);
  init_x<<<(NBONE * Bb * DD + 255) / 256, 256, 0, stream>>>(p, xall);

  const size_t bh = (size_t)Bb * HID;
  for (int f = 0; f < SSv; ++f) {
    const int pr = f & 1, nx = pr ^ 1;
    const size_t prOff = (size_t)pr * slab, nxOff = (size_t)nx * slab;
    // Layer 0: sequential bone chain
    for (int bone = 0; bone < NBONE; ++bone) {
      st_cell<1><<<32, 128, 0, stream>>>(
          WT0 + (size_t)bone * NGATE * KT0, b0 + (size_t)bone * NGATE,
          xall + ((size_t)pr * NBONE + bone) * Bb * 32,
          h0g + prOff + (size_t)(bone + 1) * bh, h0c + (size_t)bone * bh,
          c0g + prOff + (size_t)(bone + 1) * bh, c0c + (size_t)bone * bh,
          h0g + nxOff + (size_t)(bone + 1) * bh, h0c + (size_t)(bone + 1) * bh,
          c0g + nxOff + (size_t)(bone + 1) * bh, c0c + (size_t)(bone + 1) * bh,
          nullptr, 0);
    }
    // Layer 1: input x = layer-0 column h (h0c[bone]); writes pre_c to d_out
    for (int bone = 0; bone < NBONE; ++bone) {
      st_cell<16><<<32, 128, 0, stream>>>(
          WT1 + (size_t)bone * NGATE * KT1, b1 + (size_t)bone * NGATE,
          h0c + (size_t)bone * bh,
          h1g + prOff + (size_t)(bone + 1) * bh, h1c + (size_t)bone * bh,
          c1g + prOff + (size_t)(bone + 1) * bh, c1c + (size_t)bone * bh,
          h1g + nxOff + (size_t)(bone + 1) * bh, h1c + (size_t)(bone + 1) * bh,
          c1g + nxOff + (size_t)(bone + 1) * bh, c1c + (size_t)(bone + 1) * bh,
          out + PREC_BASE + ((size_t)f * NBONE + bone) * HID, SSv * NBONE * HID);
    }
    // Output projection + next-frame input
    proj<<<(Bb * NBONE * DD + 255) / 256, 256, 0, stream>>>(
        h1c, Wout, bout,
        xall + (size_t)pr * NBONE * Bb * 32,
        xall + (size_t)nx * NBONE * Bb * 32,
        out, f);
  }
}
